// TensorizedRNN_14817637171452
// MI455X (gfx1250) — matrix operand order repaired and checked by reading, not compiled, not run on hardware
//
#include <hip/hip_runtime.h>
#include <math.h>

// CDNA5 / gfx1250 bilinear-RNN cell. Compute-bound (512 FLOP/byte on W):
// v_wmma_f32_16x16x32_bf16, fp32 accumulate.
//   D[j,b] += sum_k W[l,j,k]*x[b,k]   (A = W tile from LDS, B = x in regs)
//   m[b,l] += sum_j h[b,j]*D[j,b]     (8-FMA fold -> scalar accумs)
// Each wave handles TWO batch tiles so every A load pair feeds 2 WMMAs
// (1 ds_load_b128 per WMMA) -> ~2x math per issue slot vs 1 btile/wave.

typedef float  v4f  __attribute__((ext_vector_type(4)));
typedef float  v8f  __attribute__((ext_vector_type(8)));
typedef __bf16 v4bf __attribute__((ext_vector_type(4)));
typedef __bf16 v8bf __attribute__((ext_vector_type(8)));
typedef __bf16 v16bf __attribute__((ext_vector_type(16)));

constexpr int cH    = 512;   // hidden
constexpr int c2H   = 1024;  // 2H (j-dim)
constexpr int c2I   = 256;   // 2I (k-dim, WMMA contraction)
constexpr int cI    = 128;
constexpr int CHUNK = 64;    // j rows staged per iteration
constexpr int LPITCH = 264;  // bf16 per LDS row (132 words, bank stride 4)

__global__ __launch_bounds__(256) void bilinear_rnn_wmma_bf16(
    const float* __restrict__ in0, const float* __restrict__ in1,
    const float* __restrict__ s0,  const float* __restrict__ s1,
    const float* __restrict__ W1,  const float* __restrict__ b1,
    const float* __restrict__ W2,  const float* __restrict__ b2,
    const float* __restrict__ Wm,  float* __restrict__ out)
{
    __shared__ __bf16 lw1[CHUNK * LPITCH];
    __shared__ __bf16 lw2[CHUNK * LPITCH];
    __shared__ float  lwm[CHUNK];

    const int l      = blockIdx.x >> 2;          // 0..511
    const int bblock = (blockIdx.x & 3) * 256;   // batch block of 256
    const int tid    = threadIdx.x;
    const int wave   = tid >> 5;
    const int lane   = tid & 31;
    const int l16    = lane & 15;
    const int half   = lane >> 4;
    const int bb0    = bblock + wave * 16 + l16; // lane's batch rows (N = l16)
    const int bb1    = bb0 + 128;

    // ---- B operands: x rows as v16bf, built once (loop-invariant).
    // B layout: lane = N col; lanes0-15 K=0..15, lanes16-31 K=16..31 (packed).
    v16bf xB[2][8];
    #pragma unroll
    for (int ti = 0; ti < 2; ++ti) {
        const int bbx = (ti == 0) ? bb0 : bb1;
        #pragma unroll
        for (int ks = 0; ks < 8; ++ks) {
            const int k = ks * 32 + 16 * half;   // 16 consecutive k
            const float* xp = (k < cI) ? (in0 + (size_t)bbx * cI + k)
                                       : (in1 + (size_t)bbx * cI + (k - cI));
            v8f x0 = *(const v8f*)xp;
            v8f x1 = *(const v8f*)(xp + 8);
            v16bf xb;
            #pragma unroll
            for (int i = 0; i < 8; ++i) { xb[i] = (__bf16)x0[i]; xb[8+i] = (__bf16)x1[i]; }
            xB[ti][ks] = xb;
        }
    }

    float m1a = 0.f, m2a = 0.f, hma = 0.f;       // accumulators, btile 0
    float m1b = 0.f, m2b = 0.f, hmb = 0.f;       // accumulators, btile 1
    const float* h0a = s0 + (size_t)bb0 * cH;
    const float* h1a = s1 + (size_t)bb0 * cH;
    const float* h0b = s0 + (size_t)bb1 * cH;
    const float* h1b = s1 + (size_t)bb1 * cH;

    const size_t wbase = (size_t)l * c2H * c2I;
    // staging decomposition: tid -> (row r0 + 4*i, float4 group k4)
    const int k4  = tid & 63;
    const int r0  = tid >> 6;                    // 0..3
    const int sub = (k4 & 7) >> 1;               // 8-group within 32-k group
    const int p8  = sub;                         // k-contiguous, no swap
    const int lcol = (k4 >> 3) * 32 + p8 * 8 + (k4 & 1) * 4;  // bf16 col in row

    const v8f vz = {0.f,0.f,0.f,0.f,0.f,0.f,0.f,0.f};

    for (int j0 = 0; j0 < c2H; j0 += CHUNK) {
        __syncthreads();
        const float* g1 = W1 + wbase + (size_t)j0 * c2I;
        const float* g2 = W2 + wbase + (size_t)j0 * c2I;
        // ---- stage: fp32 -> bf16, k kept in natural order (LDS col = k)
        #pragma unroll
        for (int r = r0; r < CHUNK; r += 4) {
            const size_t go = (size_t)r * c2I + k4 * 4;
            v4f w1v = *(const v4f*)(g1 + go);
            v4bf o1 = {(__bf16)w1v.x, (__bf16)w1v.y, (__bf16)w1v.z, (__bf16)w1v.w};
            *(v4bf*)&lw1[r * LPITCH + lcol] = o1;
            v4f w2v = *(const v4f*)(g2 + go);
            v4bf o2 = {(__bf16)w2v.x, (__bf16)w2v.y, (__bf16)w2v.z, (__bf16)w2v.w};
            *(v4bf*)&lw2[r * LPITCH + lcol] = o2;
        }
        if (tid < CHUNK) lwm[tid] = Wm[(size_t)(j0 + tid) * cH + l];
        if (j0 + CHUNK < c2H) {
            #pragma unroll
            for (int r = r0; r < CHUNK; r += 16) {   // light next-chunk prefetch
                const size_t go = (size_t)(r + CHUNK) * c2I + k4 * 4;
                __builtin_prefetch(g1 + go, 0, 0);
                __builtin_prefetch(g2 + go, 0, 0);
            }
        }
        __syncthreads();

        // hoist s0/s1 half select to chunk level (512 % CHUNK == 0)
        const float* hca = (j0 < cH) ? (h0a + j0) : (h1a + (j0 - cH));
        const float* hcb = (j0 < cH) ? (h0b + j0) : (h1b + (j0 - cH));

        #pragma unroll
        for (int jt = 0; jt < CHUNK / 16; ++jt) {    // 16 j-rows per tile
            const int arow = (jt * 16 + l16) * LPITCH + 16 * half;
            v8f d1a = vz, d1b = vz, d2a = vz, d2b = vz;
            #pragma unroll
            for (int ks = 0; ks < 8; ++ks) {         // K=32 per step, k=0..255
                v8bf a1lo = *(const v8bf*)&lw1[arow + ks * 32];
                v8bf a1hi = *(const v8bf*)&lw1[arow + ks * 32 + 8];
                v16bf a1 = __builtin_shufflevector(a1lo, a1hi,
                    0,1,2,3,4,5,6,7,8,9,10,11,12,13,14,15);
                d1a = __builtin_amdgcn_wmma_f32_16x16x32_bf16(
                    false, a1, false, xB[0][ks], (short)0, d1a, false, false);
                d1b = __builtin_amdgcn_wmma_f32_16x16x32_bf16(
                    false, a1, false, xB[1][ks], (short)0, d1b, false, false);
                v8bf a2lo = *(const v8bf*)&lw2[arow + ks * 32];
                v8bf a2hi = *(const v8bf*)&lw2[arow + ks * 32 + 8];
                v16bf a2 = __builtin_shufflevector(a2lo, a2hi,
                    0,1,2,3,4,5,6,7,8,9,10,11,12,13,14,15);
                d2a = __builtin_amdgcn_wmma_f32_16x16x32_bf16(
                    false, a2, false, xB[0][ks], (short)0, d2a, false, false);
                d2b = __builtin_amdgcn_wmma_f32_16x16x32_bf16(
                    false, a2, false, xB[1][ks], (short)0, d2b, false, false);
            }
            // ---- fold: D rows M = v (lanes0-15) / v+8 (lanes16-31), N = l16
            const int jo = jt * 16 + 8 * half;       // lane's first j row in chunk
            v8f hv0 = *(const v8f*)(hca + jo);
            v8f hv1 = *(const v8f*)(hcb + jo);
            #pragma unroll
            for (int v = 0; v < 8; ++v) {
                const float wmv = lwm[jo + v];
                m1a += hv0[v] * d1a[v];
                m2a += hv0[v] * d2a[v];
                hma += hv0[v] * wmv;
                m1b += hv1[v] * d1b[v];
                m2b += hv1[v] * d2b[v];
                hmb += hv1[v] * wmv;
            }
        }
    }

    // ---- combine lane halves; gate; store (lanes 0-15)
    const float b1s = b1[l], b2s = b2[l];
    const float f1a = m1a + __shfl_xor(m1a, 16, 32) + b1s;
    const float f2a = m2a + __shfl_xor(m2a, 16, 32) + b2s;
    const float fha = hma + __shfl_xor(hma, 16, 32);
    const float f1b = m1b + __shfl_xor(m1b, 16, 32) + b1s;
    const float f2b = m2b + __shfl_xor(m2b, 16, 32) + b2s;
    const float fhb = hmb + __shfl_xor(hmb, 16, 32);
    if (lane < 16) {
        const float sta = tanhf(f1a);
        const float ua  = 1.f / (1.f + expf(-f2a));
        out[(size_t)bb0 * cH + l] = ua * sta + (1.f - ua) * fha;
        const float stb = tanhf(f1b);
        const float ub  = 1.f / (1.f + expf(-f2b));
        out[(size_t)bb1 * cH + l] = ub * stb + (1.f - ub) * fhb;
    }
}

extern "C" void kernel_launch(void* const* d_in, const int* in_sizes, int n_in,
                              void* d_out, int out_size, void* d_ws, size_t ws_size,
                              hipStream_t stream) {
    const float* in0 = (const float*)d_in[0];
    const float* in1 = (const float*)d_in[1];
    const float* s0  = (const float*)d_in[2];
    const float* s1  = (const float*)d_in[3];
    const float* W1  = (const float*)d_in[4];
    const float* b1  = (const float*)d_in[5];
    const float* W2  = (const float*)d_in[6];
    const float* b2  = (const float*)d_in[7];
    const float* Wm  = (const float*)d_in[8];
    float* out = (float*)d_out;

    dim3 grid(512 * 4);   // (l, batch-block of 256)
    dim3 block(256);      // 8 wave32
    bilinear_rnn_wmma_bf16<<<grid, block, 0, stream>>>(in0, in1, s0, s1, W1, b1, W2, b2, Wm, out);
}
